// VectorQuantizer_481036337338
// MI455X (gfx1250) — matrix operand order repaired and checked by reading, not compiled, not run on hardware
//
#include <hip/hip_runtime.h>
#include <math.h>

// ---- problem constants (match reference) ----
#define DM      256        // embedding_dim
#define KCODES  1024       // num_embeddings
#define BN      65536      // 16 * 4096 rows
#define ROWS_PER_WG 128    // 8 waves * 16 rows
#define NT_TILES (KCODES / 16)   // 64 codebook tiles of 16 codes
#define TILE_BYTES (16 * DM * 2) // 8 KB per bf16 codebook tile
#define EPS_F   1.1920928955078125e-07f

typedef __attribute__((ext_vector_type(16))) __bf16 v16bf;
typedef __attribute__((ext_vector_type(8)))  float  v8f;

union FragBF {
    v16bf v;
    unsigned short u[16];
    uint4 q[2];
};

__device__ __forceinline__ unsigned short f32_to_bf16(float f) {
    unsigned int u = __float_as_uint(f);
    u += 0x7FFFu + ((u >> 16) & 1u);      // round to nearest even
    return (unsigned short)(u >> 16);
}

// CDNA5 async global->LDS copy (ASYNCcnt-tracked, no VGPR round trip).
// Each lane copies 16 bytes from its global address to its LDS offset.
__device__ __forceinline__ void async_copy_b128(unsigned lds_off,
                                                unsigned long long gaddr) {
    asm volatile("global_load_async_to_lds_b128 %0, %1, off"
                 :: "v"(lds_off), "v"(gaddr) : "memory");
}
__device__ __forceinline__ void wait_async0() {
    asm volatile("s_wait_asynccnt 0x0" ::: "memory");
}

// ---------------------------------------------------------------------------
// Phase 0: codebook f32 -> bf16, plus per-code squared norms ||e||^2
// ---------------------------------------------------------------------------
__global__ __launch_bounds__(256) void vq_prep(const float* __restrict__ E,
                                               unsigned short* __restrict__ Ebf,
                                               float* __restrict__ e2) {
    __shared__ float red[256];
    const int k = blockIdx.x;
    const int t = threadIdx.x;
    const float v = E[k * DM + t];
    Ebf[k * DM + t] = f32_to_bf16(v);
    red[t] = v * v;
    __syncthreads();
    for (int s = 128; s > 0; s >>= 1) {
        if (t < s) red[t] += red[t + s];
        __syncthreads();
    }
    if (t == 0) e2[k] = red[0];
}

// ---------------------------------------------------------------------------
// Phase 0b: zero the accumulators (counts[KCODES], sumsq)
// ---------------------------------------------------------------------------
__global__ __launch_bounds__(256) void vq_init(float* __restrict__ counts,
                                               float* __restrict__ sumsq) {
    const int t = blockIdx.x * blockDim.x + threadIdx.x;
    if (t < KCODES) counts[t] = 0.0f;
    if (t == 0) *sumsq = 0.0f;
}

// ---------------------------------------------------------------------------
// Phase 1: main kernel. 8 waves per block; wave w owns rows [row0, row0+16).
// Distances via bf16 WMMA (dual f32 accumulators), codebook tiles streamed
// into double-buffered LDS via async global->LDS copies, argmin over K,
// then f32 gather + fused MSE.
// ---------------------------------------------------------------------------
__global__ __launch_bounds__(256) void vq_main(const float* __restrict__ X,
                                               const float* __restrict__ E,
                                               const unsigned short* __restrict__ Ebf,
                                               const float* __restrict__ e2,
                                               float* __restrict__ outQ,
                                               float* __restrict__ outIdx,
                                               float* __restrict__ counts,
                                               float* __restrict__ sumsq) {
    __shared__ __align__(16) unsigned short sE[2][16 * DM];  // 2 x 8 KB tiles
    __shared__ int sIdx[ROWS_PER_WG];

    const int t    = threadIdx.x;
    const int wave = t >> 5;
    const int lane = t & 31;
    const int m    = lane & 15;    // M row within tile (A) / N col (B,C,D)
    const int half = lane >> 4;    // lane group
    const int row0 = blockIdx.x * ROWS_PER_WG + wave * 16;

    // LDS byte offset of the staging buffers (generic addr low 32 bits = LDS off)
    const unsigned sEbase = (unsigned)(unsigned long long)(void*)&sE[0][0];

    // ---- kick off async stage of tile 0 into buffer 0 ----
    {
        const unsigned long long g =
            (unsigned long long)(const void*)Ebf + (unsigned long long)t * 16u;
        async_copy_b128(sEbase + t * 16, g);
        async_copy_b128(sEbase + t * 16 + TILE_BYTES / 2, g + TILE_BYTES / 2);
    }

    // ---- load this wave's 16x256 X tile as 8 bf16 A-fragments (16x32 layout) ----
    // lane half=0: slots 0..7 -> K = k0+0..7,   slots 8..15 -> K = k0+16..23
    // lane half=1: slots 0..7 -> K = k0+8..15,  slots 8..15 -> K = k0+24..31
    FragBF a[8];
    {
        const float* xrow = X + (size_t)(row0 + m) * DM;
        #pragma unroll
        for (int kc = 0; kc < 8; ++kc) {
            const int k0 = kc * 32;
            const float4 f0 = *(const float4*)(xrow + k0 + half * 8);
            const float4 f1 = *(const float4*)(xrow + k0 + half * 8 + 4);
            const float4 f2 = *(const float4*)(xrow + k0 + 16 + half * 8);
            const float4 f3 = *(const float4*)(xrow + k0 + 16 + half * 8 + 4);
            a[kc].u[0]  = f32_to_bf16(f0.x); a[kc].u[1]  = f32_to_bf16(f0.y);
            a[kc].u[2]  = f32_to_bf16(f0.z); a[kc].u[3]  = f32_to_bf16(f0.w);
            a[kc].u[4]  = f32_to_bf16(f1.x); a[kc].u[5]  = f32_to_bf16(f1.y);
            a[kc].u[6]  = f32_to_bf16(f1.z); a[kc].u[7]  = f32_to_bf16(f1.w);
            a[kc].u[8]  = f32_to_bf16(f2.x); a[kc].u[9]  = f32_to_bf16(f2.y);
            a[kc].u[10] = f32_to_bf16(f2.z); a[kc].u[11] = f32_to_bf16(f2.w);
            a[kc].u[12] = f32_to_bf16(f3.x); a[kc].u[13] = f32_to_bf16(f3.y);
            a[kc].u[14] = f32_to_bf16(f3.z); a[kc].u[15] = f32_to_bf16(f3.w);
        }
    }

    float minv[8];
    int   mini[8];
    #pragma unroll
    for (int r = 0; r < 8; ++r) { minv[r] = 3.4e38f; mini[r] = 0; }

    for (int nt = 0; nt < NT_TILES; ++nt) {
        // tile nt was async-copied into buffer (nt&1); wait for our copies,
        // then barrier so every wave's slice of the tile is visible.
        wait_async0();
        __syncthreads();

        // prefetch tile nt+1 into the other buffer (safe: the barrier above
        // proves all waves finished computing on it last iteration)
        if (nt + 1 < NT_TILES) {
            const unsigned nxtBase = sEbase + (unsigned)((nt + 1) & 1) * TILE_BYTES;
            const unsigned long long g =
                (unsigned long long)(const void*)(Ebf + (size_t)(nt + 1) * 16 * DM)
                + (unsigned long long)t * 16u;
            async_copy_b128(nxtBase + t * 16, g);
            async_copy_b128(nxtBase + t * 16 + TILE_BYTES / 2, g + TILE_BYTES / 2);
        }

        // ---- 16x16 distance tile via WMMA, dual accumulators to break the
        //      WMMA->WMMA RAW chain ----
        const unsigned short* base = &sE[nt & 1][0];
        v8f c0 = {}, c1 = {};
        #pragma unroll
        for (int kc = 0; kc < 8; kc += 2) {
            // B 32x16 bf16 layout: half h slots 0..7 K=8h..8h+7, slots 8..15 K=16+8h..; N=lane%16
            FragBF b0, b1;
            const unsigned short* p0 = base + m * DM + kc * 32 + half * 8;
            const unsigned short* p1 = base + m * DM + (kc + 1) * 32 + half * 8;
            b0.q[0] = *(const uint4*)p0;       b0.q[1] = *(const uint4*)(p0 + 16);
            b1.q[0] = *(const uint4*)p1;       b1.q[1] = *(const uint4*)(p1 + 16);
            c0 = __builtin_amdgcn_wmma_f32_16x16x32_bf16(
                     false, a[kc].v, false, b0.v, (short)0, c0, false, false);
            c1 = __builtin_amdgcn_wmma_f32_16x16x32_bf16(
                     false, a[kc + 1].v, false, b1.v, (short)0, c1, false, false);
        }

        const float e2v  = e2[nt * 16 + m];   // N = lane%16; L2-resident
        const int   nidx = nt * 16 + m;
        #pragma unroll
        for (int r = 0; r < 8; ++r) {
            // dist - ||x||^2 = ||e||^2 - 2 x.e  (||x||^2 constant per row)
            const float d = e2v - 2.0f * (c0[r] + c1[r]);
            if (d < minv[r]) { minv[r] = d; mini[r] = nidx; }
        }
    }

    // ---- argmin reduction across the 16 N-lanes of each half (wave32 shuffles) ----
    #pragma unroll
    for (int r = 0; r < 8; ++r) {
        float v = minv[r];
        int   i = mini[r];
        #pragma unroll
        for (int mk = 1; mk < 16; mk <<= 1) {
            const float ov = __shfl_xor(v, mk, 32);
            const int   oi = __shfl_xor(i, mk, 32);
            if (ov < v || (ov == v && oi < i)) { v = ov; i = oi; }
        }
        if (m == 0) {                        // lanes 0 (M=r) and 16 (M=r+8)
            const int M = r + half * 8;
            sIdx[wave * 16 + M]  = i;
            outIdx[row0 + M]     = (float)i;
            atomicAdd(&counts[i], 1.0f);
        }
    }
    __syncthreads();

    // ---- gather quantized = emb[idx] (f32), write out, accumulate sum((q-x)^2) ----
    const int blockRow0 = blockIdx.x * ROWS_PER_WG;
    float sq = 0.0f;
    for (int e = t * 4; e < ROWS_PER_WG * DM; e += 256 * 4) {
        const int rloc = e >> 8;       // / DM
        const int col  = e & 255;
        const int code = sIdx[rloc];
        const float4 xv = *(const float4*)(X + (size_t)(blockRow0 + rloc) * DM + col);
        const float4 ev = *(const float4*)(E + (size_t)code * DM + col);
        *(float4*)(outQ + (size_t)(blockRow0 + rloc) * DM + col) = ev;
        const float dx = ev.x - xv.x, dy = ev.y - xv.y;
        const float dz = ev.z - xv.z, dw = ev.w - xv.w;
        sq += dx * dx + dy * dy + dz * dz + dw * dw;
    }
    atomicAdd(sumsq, sq);
}

// ---------------------------------------------------------------------------
// Phase 2: perplexity + loss scalars
// ---------------------------------------------------------------------------
__global__ __launch_bounds__(256) void vq_final(const float* __restrict__ counts,
                                                const float* __restrict__ sumsq,
                                                float* __restrict__ outPerp,
                                                float* __restrict__ outLoss) {
    __shared__ float red[256];
    const int t = threadIdx.x;
    float acc = 0.0f;
    for (int i = t; i < KCODES; i += 256) {
        const float p = counts[i] * (1.0f / (float)BN);
        acc += -p * logf(p + EPS_F);
    }
    red[t] = acc;
    __syncthreads();
    for (int s = 128; s > 0; s >>= 1) {
        if (t < s) red[t] += red[t + s];
        __syncthreads();
    }
    if (t == 0) {
        *outPerp = expf(red[0]);
        // q_latent == e_latent numerically -> loss = (1 + COMMIT) * mean((q-x)^2)
        *outLoss = (*sumsq) * (1.1f / ((float)BN * (float)DM));
    }
}

// ---------------------------------------------------------------------------
// launch
// ---------------------------------------------------------------------------
extern "C" void kernel_launch(void* const* d_in, const int* in_sizes, int n_in,
                              void* d_out, int out_size, void* d_ws, size_t ws_size,
                              hipStream_t stream) {
    const float* X = (const float*)d_in[0];     // [16,4096,256] f32
    const float* E = (const float*)d_in[1];     // [1024,256]    f32

    // workspace: bf16 codebook (512KB) | e2[1024] | counts[1024] | sumsq  (~533KB)
    unsigned short* Ebf = (unsigned short*)d_ws;
    float* e2     = (float*)((char*)d_ws + (size_t)KCODES * DM * sizeof(unsigned short));
    float* counts = e2 + KCODES;
    float* smsq   = counts + KCODES;

    float* outQ    = (float*)d_out;             // quantized_st  [BN*DM]
    float* outIdx  = outQ + (size_t)BN * DM;    // encoding_indices [BN] (as float)
    float* outPerp = outIdx + BN;               // perplexity scalar
    float* outLoss = outPerp + 1;               // loss_vq scalar

    vq_prep<<<KCODES, 256, 0, stream>>>(E, Ebf, e2);
    vq_init<<<(KCODES + 255) / 256, 256, 0, stream>>>(counts, smsq);
    vq_main<<<BN / ROWS_PER_WG, 256, 0, stream>>>(X, E, Ebf, e2,
                                                  outQ, outIdx, counts, smsq);
    vq_final<<<1, 256, 0, stream>>>(counts, smsq, outPerp, outLoss);
}
